// GCNModel_29540785062552
// MI455X (gfx1250) — matrix operand order repaired and checked by reading, not compiled, not run on hardware
//
#include <hip/hip_runtime.h>
#include <hip/hip_bf16.h>
#include <stdint.h>

#define NNODES 100000
#define HID    512
#define NEDGES 160000
#define BN_EPS 1e-5f
#define SLOPE  0.01f

typedef __attribute__((ext_vector_type(16))) __bf16 v16bf;
typedef __attribute__((ext_vector_type(8)))  __bf16 v8bf;
typedef __attribute__((ext_vector_type(8)))  float  v8f;

__device__ __forceinline__ unsigned short f2bf(float f) {
  unsigned int u = __float_as_uint(f);
  u += 0x7fffu + ((u >> 16) & 1u);          // round-to-nearest-even
  return (unsigned short)(u >> 16);
}

// ---------------------------------------------------------------------------
// bf16 WMMA GEMM: C[N,512] = A[N,512] @ B[512,512] (+ fused epilogue)
// A: bf16 row-major. Bt: bf16 row-major transpose of weight (Bt row = out col).
// Workgroup: 256 threads (8 waves), tile 64 rows x 512 cols; wave w -> cols [w*64,w*64+64)
// A tile staged into a double-buffered LDS tile with GLOBAL_LOAD_ASYNC_TO_LDS_B128
// (ASYNCcnt-tracked, no VGPR round trip), copy of tile k+1 overlapped with WMMAs on k.
// mode 1: out_bf = bf16(acc + bias)
// mode 2: out_f  = acc;  agg = dinv[row]^2 * acc + bc[col]   (GCN self-loop init)
// mode 3: out_f  = acc + bias
// ---------------------------------------------------------------------------
__global__ __launch_bounds__(256)
void k_gemm(const unsigned short* __restrict__ A,
            const unsigned short* __restrict__ Bt,
            const float* __restrict__ bias,
            float* __restrict__ outF,
            unsigned short* __restrict__ outB,
            const float* __restrict__ dinv,
            const float* __restrict__ bc,
            float* __restrict__ agg,
            int mode)
{
  __shared__ __align__(16) unsigned short ldsA[2][64 * 32]; // 2 x (64 rows x 32 K bf16) = 8KB
  const int tid  = threadIdx.x;
  const int lane = tid & 31;
  const int wv   = tid >> 5;
  const int h    = lane >> 4;       // half-wave select
  const int l15  = lane & 15;
  const int m0   = blockIdx.x * 64;
  const int n0   = wv * 64;

  const int sr = tid >> 2;          // staging row 0..63
  const int sc = tid & 3;           // 16B chunk 0..3 within 64B row

  const int grow = m0 + sr;
  const bool rowok = (grow < NNODES);
  // LDS byte address for async dest = low 32 bits of the flat shared pointer
  const unsigned ldsDst0 = (unsigned)(uintptr_t)(&ldsA[0][0]) + (unsigned)((sr * 4 + sc) * 16);
  const unsigned short* gpBase = A + (size_t)grow * HID + sc * 8;

  v8f acc[4][4] = {};

  // prologue: kick off async copy of k-tile 0 into buffer 0
  {
    if (rowok) {
      const unsigned short* gp = gpBase;
      asm volatile("global_load_async_to_lds_b128 %0, %1, off"
                   :: "v"(ldsDst0), "v"(gp) : "memory");
    } else {
      uint4 z = {0u, 0u, 0u, 0u};
      ((uint4*)&ldsA[0][0])[sr * 4 + sc] = z;
    }
  }

  for (int ki = 0; ki < HID / 32; ++ki) {
    const int buf = ki & 1;
    // my wave's async tile writes (issued last iteration / prologue) are done
    asm volatile("s_wait_asynccnt 0" ::: "memory");
    __syncthreads();  // everyone's tile-ki writes visible; buf^1 readers from ki-1 done

    // issue async copy of tile ki+1 into the other buffer, overlapped with compute
    if (ki + 1 < HID / 32) {
      if (rowok) {
        const unsigned short* gp = gpBase + (ki + 1) * 32;
        const unsigned ldsDst = ldsDst0 + (unsigned)((buf ^ 1) * 64 * 32 * 2);
        asm volatile("global_load_async_to_lds_b128 %0, %1, off"
                     :: "v"(ldsDst), "v"(gp) : "memory");
        if (ki + 2 < HID / 32)
          __builtin_prefetch(gp + 32, 0, 0);   // 2 tiles ahead -> global_prefetch_b8
      } else {
        uint4 z = {0u, 0u, 0u, 0u};
        ((uint4*)&ldsA[buf ^ 1][0])[sr * 4 + sc] = z;
      }
    }

    const int k0 = ki * 32;
    const unsigned short* lbase = &ldsA[buf][0];

    // --- B fragments (32x16 bf16): lane = col; lanes0-15 K 0-7 & 16-23, lanes16-31 K 8-15 & 24-31 ---
    v16bf bfrag[4];
#pragma unroll
    for (int nt = 0; nt < 4; ++nt) {
      const int col = n0 + nt * 16 + l15;
      const v8bf* bp = (const v8bf*)(Bt + (size_t)col * HID + k0 + h * 8);
      bfrag[nt] = __builtin_shufflevector(bp[0], bp[2],
          0,1,2,3,4,5,6,7,8,9,10,11,12,13,14,15);
    }
#pragma unroll
    for (int mt = 0; mt < 4; ++mt) {
      // A fragment (16x32): lane = row; lanes0-15 K 0-7 & 16-23, lanes16-31 K 8-15 & 24-31
      const v8bf* ap = (const v8bf*)(lbase + (mt * 16 + l15) * 32 + h * 8);
      const v16bf afrag = __builtin_shufflevector(ap[0], ap[2],
          0,1,2,3,4,5,6,7,8,9,10,11,12,13,14,15);
#pragma unroll
      for (int nt = 0; nt < 4; ++nt) {
        acc[mt][nt] = __builtin_amdgcn_wmma_f32_16x16x32_bf16(
            false, afrag, false, bfrag[nt], (short)0, acc[mt][nt], false, false);
      }
    }
  }

  // --- epilogue: C/D layout -> lane l15 = col, VGPR r -> row r + 8*h ---
#pragma unroll
  for (int mt = 0; mt < 4; ++mt) {
#pragma unroll
    for (int nt = 0; nt < 4; ++nt) {
      const int col = n0 + nt * 16 + l15;
#pragma unroll
      for (int r = 0; r < 8; ++r) {
        const int row = m0 + mt * 16 + h * 8 + r;
        if (row < NNODES) {
          const float v = acc[mt][nt][r];
          const size_t idx = (size_t)row * HID + col;
          if (mode == 1) {
            outB[idx] = f2bf(v + bias[col]);
          } else if (mode == 2) {
            outF[idx] = v;
            const float di = dinv[row];
            agg[idx] = di * di * v + bc[col];
          } else {
            outF[idx] = v + bias[col];
          }
        }
      }
    }
  }
}

// ---------------------------------------------------------------------------
__global__ void k_cast_bf16(const float* __restrict__ in, unsigned short* __restrict__ out, long n4) {
  const long i = (long)blockIdx.x * blockDim.x + threadIdx.x;
  if (i < n4) {
    const float4 v = ((const float4*)in)[i];
    ushort4 o;
    o.x = f2bf(v.x); o.y = f2bf(v.y); o.z = f2bf(v.z); o.w = f2bf(v.w);
    ((ushort4*)out)[i] = o;
  }
}

__global__ void k_transpose_cast(const float* __restrict__ W, unsigned short* __restrict__ Wt) {
  const int idx = blockIdx.x * 256 + threadIdx.x;      // 512*512 total
  const int k = idx >> 9, n = idx & 511;
  Wt[(size_t)n * HID + k] = f2bf(W[(size_t)k * HID + n]);
}

__global__ void k_zero(float* __restrict__ p, long n) {
  const long i = (long)blockIdx.x * blockDim.x + threadIdx.x;
  if (i < n) p[i] = 0.f;
}

__global__ void k_deg(const int* __restrict__ dst, float* __restrict__ deg) {
  const int e = blockIdx.x * 256 + threadIdx.x;
  if (e < NEDGES) atomicAdd(&deg[dst[e]], 1.0f);
}

__global__ void k_dinv(const float* __restrict__ deg, float* __restrict__ dinv) {
  const int i = blockIdx.x * 256 + threadIdx.x;
  if (i < NNODES) dinv[i] = rsqrtf(deg[i] + 1.0f);     // +1 self loop
}

// edge messages: agg[dst] += dinv[src]*dinv[dst] * t[src]; 128 threads/edge, 4 feats/thread
__global__ __launch_bounds__(256)
void k_edge(const int* __restrict__ src, const int* __restrict__ dst,
            const float* __restrict__ dinv, const float* __restrict__ t,
            float* __restrict__ agg) {
  const long g = (long)blockIdx.x * 256 + threadIdx.x;
  const int e  = (int)(g >> 7);
  const int c4 = (int)(g & 127) * 4;
  if (e < NEDGES) {
    const int s = src[e], d = dst[e];
    const float w = dinv[s] * dinv[d];
    const float4 v = *(const float4*)(t + (size_t)s * HID + c4);
    float* ap = agg + (size_t)d * HID + c4;
    atomicAdd(ap + 0, w * v.x);
    atomicAdd(ap + 1, w * v.y);
    atomicAdd(ap + 2, w * v.z);
    atomicAdd(ap + 3, w * v.w);
  }
}

// per-feature partial sums/sumsq over a 256-row block (coalesced), atomic into globals
__global__ __launch_bounds__(256)
void k_bn_stats(const float* __restrict__ agg, float* __restrict__ sums,
                float* __restrict__ sumsq) {
  const int f  = threadIdx.x;
  const int r0 = blockIdx.x * 256;
  const int re = (r0 + 256 < NNODES) ? r0 + 256 : NNODES;
  float s0 = 0.f, q0 = 0.f, s1 = 0.f, q1 = 0.f;
  for (int r = r0; r < re; ++r) {
    const float* rowp = agg + (size_t)r * HID;
    const float v0 = rowp[f], v1 = rowp[f + 256];
    s0 += v0; q0 += v0 * v0;
    s1 += v1; q1 += v1 * v1;
  }
  atomicAdd(&sums[f], s0);        atomicAdd(&sumsq[f], q0);
  atomicAdd(&sums[f + 256], s1);  atomicAdd(&sumsq[f + 256], q1);
}

__global__ void k_bn_final(const float* __restrict__ sums, const float* __restrict__ sumsq,
                           const float* __restrict__ gamma, const float* __restrict__ beta,
                           float* __restrict__ scale, float* __restrict__ shift) {
  const int f = threadIdx.x;     // 512 threads, 1 block
  const float inv_n = 1.0f / (float)NNODES;
  const float mu  = sums[f] * inv_n;
  const float var = sumsq[f] * inv_n - mu * mu;
  const float sc  = gamma[f] * rsqrtf(var + BN_EPS);
  scale[f] = sc;
  shift[f] = beta[f] - mu * sc;
}

// h_bf16 = bf16(leakyrelu(agg*scale + shift))
__global__ void k_bn_apply(const float* __restrict__ agg, const float* __restrict__ scale,
                           const float* __restrict__ shift, unsigned short* __restrict__ hb,
                           long n4) {
  const long i = (long)blockIdx.x * blockDim.x + threadIdx.x;
  if (i < n4) {
    const float4 v = ((const float4*)agg)[i];
    const int c = (int)((i * 4) & (HID - 1));
    float o0 = v.x * scale[c + 0] + shift[c + 0];
    float o1 = v.y * scale[c + 1] + shift[c + 1];
    float o2 = v.z * scale[c + 2] + shift[c + 2];
    float o3 = v.w * scale[c + 3] + shift[c + 3];
    o0 = (o0 >= 0.f) ? o0 : SLOPE * o0;
    o1 = (o1 >= 0.f) ? o1 : SLOPE * o1;
    o2 = (o2 >= 0.f) ? o2 : SLOPE * o2;
    o3 = (o3 >= 0.f) ? o3 : SLOPE * o3;
    ushort4 r;
    r.x = f2bf(o0); r.y = f2bf(o1); r.z = f2bf(o2); r.w = f2bf(o3);
    ((ushort4*)hb)[i] = r;
  }
}

// out[row] = sigmoid(h2[row,:] . WO + bO); one wave32 per row, 8 rows per block
__global__ __launch_bounds__(256)
void k_final(const float* __restrict__ h2, const float* __restrict__ WO,
             const float* __restrict__ bO, float* __restrict__ out) {
  const int lane = threadIdx.x & 31;
  const int wv   = threadIdx.x >> 5;
  const int row  = blockIdx.x * 8 + wv;
  if (row < NNODES) {
    const float* hp = h2 + (size_t)row * HID;
    float s = 0.f;
    for (int k = lane; k < HID; k += 32) s += hp[k] * WO[k];
    for (int off = 16; off > 0; off >>= 1) s += __shfl_xor(s, off, 32);
    if (lane == 0) out[row] = 1.0f / (1.0f + __expf(-(s + bO[0])));
  }
}

// ---------------------------------------------------------------------------
extern "C" void kernel_launch(void* const* d_in, const int* in_sizes, int n_in,
                              void* d_out, int out_size, void* d_ws, size_t ws_size,
                              hipStream_t stream) {
  const float* x     = (const float*)d_in[0];
  const int*   ei    = (const int*)d_in[1];
  const float* W1    = (const float*)d_in[2];
  const float* b1    = (const float*)d_in[3];
  const float* Wc    = (const float*)d_in[4];
  const float* bc    = (const float*)d_in[5];
  const float* gamma = (const float*)d_in[6];
  const float* beta  = (const float*)d_in[7];
  const float* W2    = (const float*)d_in[8];
  const float* b2    = (const float*)d_in[9];
  const float* WO    = (const float*)d_in[10];
  const float* bO    = (const float*)d_in[11];
  float* out = (float*)d_out;

  const int E = in_sizes[1] / 2;
  const int* src = ei;
  const int* dst = ei + E;
  (void)E; (void)n_in; (void)out_size; (void)ws_size;

  // workspace carve-up
  char* w = (char*)d_ws;
  size_t off = 0;
  auto alloc = [&](size_t bytes) -> char* {
    char* p = w + off;
    off += (bytes + 255) & ~(size_t)255;
    return p;
  };
  const size_t NB = (size_t)NNODES * HID;
  unsigned short* xb   = (unsigned short*)alloc(NB * 2);
  unsigned short* hbf  = (unsigned short*)alloc(NB * 2);
  float*          t    = (float*)alloc(NB * 4);          // GEMM2 out / reused as h2
  float*          agg  = (float*)alloc(NB * 4);
  unsigned short* W1t  = (unsigned short*)alloc(512 * 512 * 2);
  unsigned short* Wct  = (unsigned short*)alloc(512 * 512 * 2);
  unsigned short* W2t  = (unsigned short*)alloc(512 * 512 * 2);
  float*          deg  = (float*)alloc(NNODES * 4);
  float*          dinv = (float*)alloc(NNODES * 4);
  float*          sums = (float*)alloc(1024 * 4);        // [0,512)=sum, [512,1024)=sumsq
  float*          scl  = (float*)alloc(1024 * 4);        // [0,512)=scale, [512,1024)=shift

  const long n4 = (long)NB / 4;                          // float4 / ushort4 elements
  const int  cblk = (int)((n4 + 255) / 256);
  const int  gblocks = (NNODES + 63) / 64;               // 1563 GEMM workgroups

  // precompute
  k_cast_bf16<<<cblk, 256, 0, stream>>>(x, xb, n4);
  k_transpose_cast<<<1024, 256, 0, stream>>>(W1, W1t);
  k_transpose_cast<<<1024, 256, 0, stream>>>(Wc, Wct);
  k_transpose_cast<<<1024, 256, 0, stream>>>(W2, W2t);
  k_zero<<<(NNODES + 255) / 256, 256, 0, stream>>>(deg, NNODES);
  k_deg<<<(NEDGES + 255) / 256, 256, 0, stream>>>(dst, deg);
  k_dinv<<<(NNODES + 255) / 256, 256, 0, stream>>>(deg, dinv);

  // linear1: h = bf16(x @ W1 + b1)
  k_gemm<<<gblocks, 256, 0, stream>>>(xb, W1t, b1, nullptr, hbf,
                                      nullptr, nullptr, nullptr, 1);

  // two GCNConv + BN + LeakyReLU rounds (shared params)
  for (int it = 0; it < 2; ++it) {
    // t = h @ Wc ; agg = dinv^2 * t + bc  (self-loop + bias fused)
    k_gemm<<<gblocks, 256, 0, stream>>>(hbf, Wct, nullptr, t, nullptr,
                                        dinv, bc, agg, 2);
    // agg[dst] += dinv[src]*dinv[dst]*t[src]
    k_edge<<<(NEDGES * 128) / 256, 256, 0, stream>>>(src, dst, dinv, t, agg);
    // batchnorm
    k_zero<<<4, 256, 0, stream>>>(sums, 1024);
    k_bn_stats<<<(NNODES + 255) / 256, 256, 0, stream>>>(agg, sums, sums + 512);
    k_bn_final<<<1, 512, 0, stream>>>(sums, sums + 512, gamma, beta, scl, scl + 512);
    // h = bf16(leakyrelu(bn(agg)))
    k_bn_apply<<<cblk, 256, 0, stream>>>(agg, scl, scl + 512, hbf, n4);
  }

  // linear2: h2 = h @ W2 + b2  (f32, reuse t buffer)
  k_gemm<<<gblocks, 256, 0, stream>>>(hbf, W2t, b2, t, nullptr,
                                      nullptr, nullptr, nullptr, 3);
  // out = sigmoid(h2 @ WO + bO)
  k_final<<<(NNODES + 7) / 8, 256, 0, stream>>>(t, WO, bO, out);
}
